// MultiHeadAttentionBlock_41489384079516
// MI455X (gfx1250) — matrix operand order repaired and checked by reading, not compiled, not run on hardware
//
#include <hip/hip_runtime.h>

// ---------------------------------------------------------------------------
// MHA block for MI455X (gfx1250, wave32, WMMA 16x16x32 f16)
// B=4, S=2048, D=512, H=8, HD=64
//
// Pipeline:
//   0) cvt:       f32 -> f16 copies of activations + weights (one pass)
//   1) qkv_proj:  pure-f16 WMMA GEMM, Q/K -> (B,H,S,HD), V -> (B,H,HD,S)^T
//   2) flash_attn causal online-softmax, 8 WMMA per 32-key chunk
//   3) out_proj:  f16 WMMA GEMM + bias, f32 out
// ---------------------------------------------------------------------------

typedef __attribute__((ext_vector_type(16))) _Float16     v16h;
typedef __attribute__((ext_vector_type(8)))  float        v8f;
typedef __attribute__((ext_vector_type(8)))  unsigned int v8u;
typedef _Float16 half_t;

constexpr int Bc  = 4;
constexpr int Sc  = 2048;
constexpr int Dc  = 512;
constexpr int Hc  = 8;
constexpr int HDc = 64;

// ---- WMMA wrapper ----------------------------------------------------------
__device__ __forceinline__ v8f wmma_f16(v16h a, v16h b, v8f c) {
  // (neg_a, A, neg_b, B, c_mod, C, reuse_a, reuse_b)
  return __builtin_amdgcn_wmma_f32_16x16x32_f16(false, a, false, b,
                                                (short)0, c, false, false);
}

// ---- Fragment loaders (f16 sources, wide b128 loads) -----------------------
// A-matrix 16x32 (MxK): lane r=l&15 -> row r; VGPR v pair:
//   k = (v<4 ? 2v : 8+2v) + 8*(l>>4)  => two contiguous 8-half runs per lane
__device__ __forceinline__ v16h load_a_h(const half_t* base, int ld, int lane) {
  const int hf = lane >> 4, r = lane & 15;
  const half_t* rowp = base + (size_t)r * ld + 8 * hf;
  const uint4 u0 = *(const uint4*)(rowp);        // k = 8hf + 0..7
  const uint4 u1 = *(const uint4*)(rowp + 16);   // k = 16 + 8hf + 0..7
  v8u t;
  t[0] = u0.x; t[1] = u0.y; t[2] = u0.z; t[3] = u0.w;
  t[4] = u1.x; t[5] = u1.y; t[6] = u1.z; t[7] = u1.w;
  return __builtin_bit_cast(v16h, t);
}

// B-matrix 32x16 (KxN) from row-major [N x K] ("BT"): lane n=l&15 -> col n;
// VGPR v pair: k = (v<4 ? 2v : 8+2v) + 8*(l>>4)  => two contiguous 8-half runs per lane
__device__ __forceinline__ v16h load_b_h(const half_t* bt, int ld, int lane) {
  const int hf = lane >> 4, n = lane & 15;
  const half_t* colp = bt + (size_t)n * ld + 8 * hf;
  const uint4 u0 = *(const uint4*)(colp);
  const uint4 u1 = *(const uint4*)(colp + 16);
  v8u t;
  t[0] = u0.x; t[1] = u0.y; t[2] = u0.z; t[3] = u0.w;
  t[4] = u1.x; t[5] = u1.y; t[6] = u1.z; t[7] = u1.w;
  return __builtin_bit_cast(v16h, t);
}

// ---------------------------------------------------------------------------
// Kernel 0: f32 -> f16 conversion (vectorized, n multiple of 4)
// ---------------------------------------------------------------------------
__global__ __launch_bounds__(256) void cvt_f16_kernel(
    const float* __restrict__ src, half_t* __restrict__ dst, int n)
{
  const int i = (blockIdx.x * 256 + threadIdx.x) * 4;
  if (i < n) {
    const float4 f = *(const float4*)(src + i);
    dst[i]     = (half_t)f.x;
    dst[i + 1] = (half_t)f.y;
    dst[i + 2] = (half_t)f.z;
    dst[i + 3] = (half_t)f.w;
  }
}

// ---------------------------------------------------------------------------
// Kernel 1: fused QKV projection.  Y = X @ W^T + bias, all-f16 WMMA inputs.
// z = 0/1/2 -> Q/K/V.  Q,K stored as (B,H,S,HD); V stored transposed
// (B,H,HD,S) so the PV B-fragments in the attention kernel load contiguously.
// Block: 256 threads = 8 waves (4 along M x 2 along N); wave tile 16x64;
// block tile 64x128.  grid = (BS/64, D/128, 3)
// ---------------------------------------------------------------------------
__global__ __launch_bounds__(256) void qkv_proj_kernel(
    const half_t* __restrict__ xq, const half_t* __restrict__ xk,
    const half_t* __restrict__ xv,
    const half_t* __restrict__ wq, const half_t* __restrict__ wk,
    const half_t* __restrict__ wv,
    const float* __restrict__ bq, const float* __restrict__ bk,
    const float* __restrict__ bv,
    half_t* __restrict__ Qh, half_t* __restrict__ Kh, half_t* __restrict__ Vt)
{
  const int z     = blockIdx.z;
  const half_t* X = (z == 0) ? xq : (z == 1) ? xk : xv;
  const half_t* W = (z == 0) ? wq : (z == 1) ? wk : wv;
  const float* bias = (z == 0) ? bq : (z == 1) ? bk : bv;

  const int wave = threadIdx.x >> 5;
  const int lane = threadIdx.x & 31;
  const int row0 = blockIdx.x * 64 + (wave >> 1) * 16;
  const int col0 = blockIdx.y * 128 + (wave & 1) * 64;

  v8f acc[4] = {};
#pragma unroll 4
  for (int k0 = 0; k0 < Dc; k0 += 32) {
    __builtin_prefetch(X + (size_t)row0 * Dc + k0 + 64, 0, 1);
    const v16h a = load_a_h(X + (size_t)row0 * Dc + k0, Dc, lane);
#pragma unroll
    for (int t = 0; t < 4; ++t) {
      const v16h b = load_b_h(W + (size_t)(col0 + 16 * t) * Dc + k0, Dc, lane);
      acc[t] = wmma_f16(a, b, acc[t]);
    }
  }

  // C layout: lane l holds column n=l&15; VGPR j holds row m = j + 8*(l>>4)
  const int hf = lane >> 4, n = lane & 15;
#pragma unroll
  for (int t = 0; t < 4; ++t) {
    const int col = col0 + 16 * t + n;
    const int h = col >> 6, d = col & 63;
    const float bcol = bias[col];
#pragma unroll
    for (int j = 0; j < 8; ++j) {
      const int row = row0 + j + 8 * hf;           // 0 .. B*S-1
      const int b_  = row >> 11, s = row & (Sc - 1);
      const half_t val = (half_t)(acc[t][j] + bcol);
      if (z == 0)
        Qh[(((size_t)(b_ * Hc + h) * Sc) + s) * HDc + d] = val;
      else if (z == 1)
        Kh[(((size_t)(b_ * Hc + h) * Sc) + s) * HDc + d] = val;
      else
        Vt[(((size_t)(b_ * Hc + h) * HDc) + d) * Sc + s] = val;   // transposed
    }
  }
}

// ---------------------------------------------------------------------------
// Kernel 2: causal flash attention.  One wave owns a 16-row query tile.
// Per 32-key chunk: 4 WMMAs for S = Q K^T, online softmax in f32, P routed
// through a private LDS tile (C-layout -> A-layout transpose), 4 WMMAs for
// O += P V.  Causal chunk skipping: only kc < q0+16 chunks are visited.
// grid = (B*H, S/128), block = 256 (8 waves).
// ---------------------------------------------------------------------------
__global__ __launch_bounds__(256) void flash_attn_kernel(
    const half_t* __restrict__ Qh, const half_t* __restrict__ Kh,
    const half_t* __restrict__ Vt, half_t* __restrict__ Xat)
{
  __shared__ half_t plds[8 * 16 * 32];   // 1 KB per wave, 8 KB total

  const int bh   = blockIdx.x;                 // b*H + h
  const int wave = threadIdx.x >> 5;
  const int lane = threadIdx.x & 31;
  const int q0   = blockIdx.y * 128 + wave * 16;

  const half_t* Qb = Qh + (size_t)bh * Sc * HDc;
  const half_t* Kb = Kh + (size_t)bh * Sc * HDc;
  const half_t* Vb = Vt + (size_t)bh * HDc * Sc;
  half_t* myp = plds + wave * (16 * 32);

  const int hf = lane >> 4, nl = lane & 15;

  // Q A-fragments for the full K=64 head dim (kept resident)
  const v16h qa0 = load_a_h(Qb + (size_t)q0 * HDc + 0,  HDc, lane);
  const v16h qa1 = load_a_h(Qb + (size_t)q0 * HDc + 32, HDc, lane);

  v8f o[4] = {};
  float m_i[8], l_i[8];
#pragma unroll
  for (int j = 0; j < 8; ++j) { m_i[j] = -1e30f; l_i[j] = 0.0f; }

  const int kend = q0 + 16;                    // exclusive key bound (causal)
  for (int kc = 0; kc < kend; kc += 32) {
    // ---- scores: two 16x16 tiles over keys [kc, kc+32) -------------------
    v8f s0 = {}, s1 = {};
    {
      v16h kb;
      kb = load_b_h(Kb + (size_t)kc * HDc + 0,         HDc, lane); s0 = wmma_f16(qa0, kb, s0);
      kb = load_b_h(Kb + (size_t)kc * HDc + 32,        HDc, lane); s0 = wmma_f16(qa1, kb, s0);
      kb = load_b_h(Kb + (size_t)(kc + 16) * HDc + 0,  HDc, lane); s1 = wmma_f16(qa0, kb, s1);
      kb = load_b_h(Kb + (size_t)(kc + 16) * HDc + 32, HDc, lane); s1 = wmma_f16(qa1, kb, s1);
    }
    __builtin_prefetch(Kb + (size_t)(kc + 32) * HDc, 0, 1);

    // ---- scale + causal mask + row max ----------------------------------
    float p0[8], p1[8], rmax[8], rsum[8], fac[8];
#pragma unroll
    for (int j = 0; j < 8; ++j) {
      const int rowg = q0 + j + 8 * hf;
      const int c0 = kc + nl, c1 = kc + 16 + nl;
      p0[j] = (c0 <= rowg) ? s0[j] * 0.125f : -1e30f;   // 1/sqrt(64)
      p1[j] = (c1 <= rowg) ? s1[j] * 0.125f : -1e30f;
      rmax[j] = fmaxf(p0[j], p1[j]);
    }
#pragma unroll
    for (int off = 1; off < 16; off <<= 1)
#pragma unroll
      for (int j = 0; j < 8; ++j)
        rmax[j] = fmaxf(rmax[j], __shfl_xor(rmax[j], off, 32));

    // ---- online softmax update ------------------------------------------
#pragma unroll
    for (int j = 0; j < 8; ++j) {
      const float newm = fmaxf(m_i[j], rmax[j]);
      fac[j] = __expf(m_i[j] - newm);
      m_i[j] = newm;
      p0[j]  = __expf(p0[j] - newm);
      p1[j]  = __expf(p1[j] - newm);
      rsum[j] = p0[j] + p1[j];
    }
#pragma unroll
    for (int off = 1; off < 16; off <<= 1)
#pragma unroll
      for (int j = 0; j < 8; ++j)
        rsum[j] += __shfl_xor(rsum[j], off, 32);
#pragma unroll
    for (int j = 0; j < 8; ++j) l_i[j] = l_i[j] * fac[j] + rsum[j];
#pragma unroll
    for (int t = 0; t < 4; ++t)
#pragma unroll
      for (int j = 0; j < 8; ++j) o[t][j] *= fac[j];

    // ---- P: C-layout -> LDS -> A-layout ---------------------------------
#pragma unroll
    for (int j = 0; j < 8; ++j) {
      const int m = j + 8 * hf;
      myp[m * 32 + nl]      = (half_t)p0[j];
      myp[m * 32 + 16 + nl] = (half_t)p1[j];
    }
    __builtin_amdgcn_wave_barrier();
    asm volatile("s_wait_dscnt 0" ::: "memory");   // DS ops in-order per wave
    const v16h pa = load_a_h(myp, 32, lane);
    __builtin_amdgcn_wave_barrier();

    // ---- O += P * V  (V stored transposed: Vt[d, s]) --------------------
#pragma unroll
    for (int t = 0; t < 4; ++t) {
      const v16h vb = load_b_h(Vb + (size_t)(16 * t) * Sc + kc, Sc, lane);
      o[t] = wmma_f16(pa, vb, o[t]);
    }
  }

  // ---- finalize & store as (B, S, D) f16 for the output projection ------
  const int b_ = bh >> 3, h = bh & 7;
#pragma unroll
  for (int j = 0; j < 8; ++j) {
    const float inv = 1.0f / l_i[j];
    const int s = q0 + j + 8 * hf;
#pragma unroll
    for (int t = 0; t < 4; ++t) {
      const int col = h * HDc + 16 * t + nl;
      Xat[((size_t)b_ * Sc + s) * Dc + col] = (half_t)(o[t][j] * inv);
    }
  }
}

// ---------------------------------------------------------------------------
// Kernel 3: output projection  Y = Xat @ Wo^T + bo  (f32 out).
// Same wave-tiling as kernel 1.  grid = (BS/64, D/128)
// ---------------------------------------------------------------------------
__global__ __launch_bounds__(256) void out_proj_kernel(
    const half_t* __restrict__ Xat, const half_t* __restrict__ wo,
    const float* __restrict__ bo, float* __restrict__ Y)
{
  const int wave = threadIdx.x >> 5;
  const int lane = threadIdx.x & 31;
  const int row0 = blockIdx.x * 64 + (wave >> 1) * 16;
  const int col0 = blockIdx.y * 128 + (wave & 1) * 64;

  v8f acc[4] = {};
#pragma unroll 4
  for (int k0 = 0; k0 < Dc; k0 += 32) {
    __builtin_prefetch(Xat + (size_t)row0 * Dc + k0 + 64, 0, 1);
    const v16h a = load_a_h(Xat + (size_t)row0 * Dc + k0, Dc, lane);
#pragma unroll
    for (int t = 0; t < 4; ++t) {
      const v16h b = load_b_h(wo + (size_t)(col0 + 16 * t) * Dc + k0, Dc, lane);
      acc[t] = wmma_f16(a, b, acc[t]);
    }
  }

  const int hf = lane >> 4, n = lane & 15;
#pragma unroll
  for (int t = 0; t < 4; ++t) {
    const int col = col0 + 16 * t + n;
    const float bcol = bo[col];
#pragma unroll
    for (int j = 0; j < 8; ++j) {
      const int row = row0 + j + 8 * hf;
      Y[(size_t)row * Dc + col] = acc[t][j] + bcol;
    }
  }
}

// ---------------------------------------------------------------------------
extern "C" void kernel_launch(void* const* d_in, const int* in_sizes, int n_in,
                              void* d_out, int out_size, void* d_ws,
                              size_t ws_size, hipStream_t stream)
{
  (void)in_sizes; (void)n_in; (void)out_size; (void)ws_size;

  const float* q  = (const float*)d_in[0];
  const float* k  = (const float*)d_in[1];
  const float* v  = (const float*)d_in[2];
  /* d_in[3] = causal mask (implicit in kernel) */
  const float* wq = (const float*)d_in[4];
  const float* bq = (const float*)d_in[5];
  const float* wk = (const float*)d_in[6];
  const float* bk = (const float*)d_in[7];
  const float* wv = (const float*)d_in[8];
  const float* bv = (const float*)d_in[9];
  const float* wo = (const float*)d_in[10];
  const float* bo = (const float*)d_in[11];
  float* out = (float*)d_out;

  // workspace layout (f16 elements):
  //   xq,xk,xv : activations (B*S*D each)
  //   wqh..woh : weights (D*D each)
  //   Qh,Kh,Vt,Xat : intermediates (B*S*D each)
  const size_t NE = (size_t)Bc * Sc * Dc;          // 4,194,304
  const size_t WE = (size_t)Dc * Dc;               //   262,144
  half_t* xqh = (half_t*)d_ws;
  half_t* xkh = xqh + NE;
  half_t* xvh = xkh + NE;
  half_t* wqh = xvh + NE;
  half_t* wkh = wqh + WE;
  half_t* wvh = wkh + WE;
  half_t* woh = wvh + WE;
  half_t* Qh  = woh + WE;
  half_t* Kh  = Qh + NE;
  half_t* Vt  = Kh + NE;
  half_t* Xat = Vt + NE;

  const int BS = Bc * Sc;                          // 8192 rows
  dim3 blk(256);

  // 0) conversions to f16 (one pass; removes all cvt from GEMM inner loops)
  const int gx_x = (int)(NE / 4 / 256);            // 4096 blocks
  const int gx_w = (int)(WE / 4 / 256);            //  256 blocks
  hipLaunchKernelGGL(cvt_f16_kernel, dim3(gx_x), blk, 0, stream, q,  xqh, (int)NE);
  hipLaunchKernelGGL(cvt_f16_kernel, dim3(gx_x), blk, 0, stream, k,  xkh, (int)NE);
  hipLaunchKernelGGL(cvt_f16_kernel, dim3(gx_x), blk, 0, stream, v,  xvh, (int)NE);
  hipLaunchKernelGGL(cvt_f16_kernel, dim3(gx_w), blk, 0, stream, wq, wqh, (int)WE);
  hipLaunchKernelGGL(cvt_f16_kernel, dim3(gx_w), blk, 0, stream, wk, wkh, (int)WE);
  hipLaunchKernelGGL(cvt_f16_kernel, dim3(gx_w), blk, 0, stream, wv, wvh, (int)WE);
  hipLaunchKernelGGL(cvt_f16_kernel, dim3(gx_w), blk, 0, stream, wo, woh, (int)WE);

  // 1) QKV projection
  dim3 g_proj(BS / 64, Dc / 128, 3);               // (128, 4, 3)
  hipLaunchKernelGGL(qkv_proj_kernel, g_proj, blk, 0, stream,
                     xqh, xkh, xvh, wqh, wkh, wvh, bq, bk, bv, Qh, Kh, Vt);

  // 2) causal flash attention
  dim3 g_attn(Bc * Hc, Sc / 128);                  // (32, 16)
  hipLaunchKernelGGL(flash_attn_kernel, g_attn, blk, 0, stream,
                     Qh, Kh, Vt, Xat);

  // 3) output projection
  dim3 g_out(BS / 64, Dc / 128);                   // (128, 4)
  hipLaunchKernelGGL(out_proj_kernel, g_out, blk, 0, stream,
                     Xat, woh, bo, out);
}
